// CardHistorySAGE_18588618457111
// MI455X (gfx1250) — matrix operand order repaired and checked by reading, not compiled, not run on hardware
//
#include <hip/hip_runtime.h>
#include <hip/hip_bf16.h>

typedef __bf16 bf16_t;
typedef __attribute__((ext_vector_type(4)))  __bf16 v4bf;
typedef __attribute__((ext_vector_type(8)))  __bf16 v8bf;
typedef __attribute__((ext_vector_type(16))) __bf16 v16bf;
typedef __attribute__((ext_vector_type(8)))  float  v8f;

// ---------------------------------------------------------------------------
// One wave computes a 16-row x 128-col tile: acc += sX[m0:m0+16, 0:K] @ W[0:K,0:128]
// with V_WMMA_F32_16X16X32_BF16 (bf16 inputs, f32 accumulate).
//
// A 16x32 bf16 operand as loaded here: lane = hl*16 + l16 holds row m0+l16 with
//   elems 0..7  = K = k0 + 16*hl + (0..7)
//   elems 8..15 = K = k0 + 16*hl + 8 + (0..7)
// B 32x16 bf16 operand as loaded here, same k order as A: lane holds col n0+l16 with
//   K = k0 + 16*hl + (0..15), 2 bf16 per VGPR, K ascending.
// Wp is pre-packed so that a lane's 16 B values are contiguous:
//   Wp[((k0/16 + hl)*128 + n)*16 + j] = bf16(W[k0 + 16*hl + j][n])
// C/D f32 16x16: VGPR i -> row m0 + i + 8*hl, col nt*16 + l16.
// ---------------------------------------------------------------------------
__device__ __forceinline__ void wave_gemm_bf16(const bf16_t* __restrict__ sX, int ldx,
                                               int m0, int K,
                                               const bf16_t* __restrict__ Wp,
                                               v8f acc[8]) {
  const int lane = threadIdx.x & 31;
  const int hl   = lane >> 4;
  const int l16  = lane & 15;
  const v8f zero = {0.f, 0.f, 0.f, 0.f, 0.f, 0.f, 0.f, 0.f};
#pragma unroll
  for (int nt = 0; nt < 8; ++nt) acc[nt] = zero;

  const bf16_t* xrow = sX + (m0 + l16) * ldx;
  for (int k0 = 0; k0 < K; k0 += 32) {
    v8bf alo = *(const v8bf*)(xrow + k0 + 16 * hl);
    v8bf ahi = *(const v8bf*)(xrow + k0 + 16 * hl + 8);
    v16bf a;
#pragma unroll
    for (int i = 0; i < 8; ++i) { a[i] = alo[i]; a[i + 8] = ahi[i]; }
    const bf16_t* wp = Wp + ((size_t)(k0 / 16 + hl) * 128) * 16;
#pragma unroll
    for (int nt = 0; nt < 8; ++nt) {
      v16bf b = *(const v16bf*)(wp + (size_t)(nt * 16 + l16) * 16);  // 32B contiguous
      acc[nt] = __builtin_amdgcn_wmma_f32_16x16x32_bf16(
          /*neg_a=*/false, a, /*neg_b=*/false, b,
          /*c_mod=*/(short)0, acc[nt], /*reuse_a=*/false, /*reuse_b=*/false);
    }
  }
}

// ---------------------------------------------------------------------------
// Pack W [realK][128] f32 -> Wp [padK/16][128][16] bf16 (zero rows past realK).
// ---------------------------------------------------------------------------
__global__ void pack_w_kernel(const float* __restrict__ W, bf16_t* __restrict__ Wp,
                              int realK, int padK) {
  const int i = blockIdx.x * blockDim.x + threadIdx.x;  // over padK*128 elements
  if (i >= padK * 128) return;
  const int j = i & 15;
  const int n = (i >> 4) & 127;
  const int p = i >> 11;
  const int k = p * 16 + j;
  const float v = (k < realK) ? W[(size_t)k * 128 + n] : 0.f;
  Wp[i] = (bf16_t)v;
}

// ---------------------------------------------------------------------------
// Kernel 1: 2-layer txn MLP (64 -> 128 -> 128, ReLU). 128 threads = 4 waves,
// 64 rows/block. HIST=false: write target_h. HIST=true: f32 atomic scatter-add.
// ---------------------------------------------------------------------------
template <bool HIST>
__global__ __launch_bounds__(128) void txn_mlp_kernel(
    const float* __restrict__ X,
    const bf16_t* __restrict__ Wp1, const float* __restrict__ b1,
    const bf16_t* __restrict__ Wp2, const float* __restrict__ b2,
    float* __restrict__ out_h,
    float* __restrict__ seg_sum, float* __restrict__ seg_cnt,
    const int* __restrict__ seg_idx) {
  __shared__ bf16_t sBuf[64 * 128];  // 16 KB: X tile (64x64 bf16) then H1 (64x128 bf16)
  const int tid  = threadIdx.x;
  const int row0 = blockIdx.x * 64;

  // Stage contiguous 64x64 f32 tile -> bf16.
  {
    const float4* src = (const float4*)(X + (size_t)row0 * 64);
    for (int i = tid; i < 64 * 64 / 4; i += 128) {
      float4 f = src[i];
      v4bf o = {(bf16_t)f.x, (bf16_t)f.y, (bf16_t)f.z, (bf16_t)f.w};
      *(v4bf*)(sBuf + (size_t)i * 4) = o;
    }
  }
  __syncthreads();

  const int lane = tid & 31;
  const int hl   = lane >> 4;
  const int l16  = lane & 15;
  const int m0   = (tid >> 5) * 16;

  v8f acc[8];
  wave_gemm_bf16(sBuf, 64, m0, 64, Wp1, acc);

  bf16_t h1[8][8];
#pragma unroll
  for (int nt = 0; nt < 8; ++nt) {
    const float bb = b1[nt * 16 + l16];
#pragma unroll
    for (int i = 0; i < 8; ++i) {
      float v = acc[nt][i] + bb;
      h1[nt][i] = (bf16_t)(v > 0.f ? v : 0.f);
    }
  }
  __syncthreads();  // everyone done reading the X tile
#pragma unroll
  for (int nt = 0; nt < 8; ++nt) {
    const int n = nt * 16 + l16;
#pragma unroll
    for (int i = 0; i < 8; ++i) sBuf[(m0 + 8 * hl + i) * 128 + n] = h1[nt][i];
  }
  __syncthreads();

  wave_gemm_bf16(sBuf, 128, m0, 128, Wp2, acc);

  int cidx[8];
  if (HIST) {
#pragma unroll
    for (int i = 0; i < 8; ++i) cidx[i] = seg_idx[row0 + m0 + 8 * hl + i];
  }
#pragma unroll
  for (int nt = 0; nt < 8; ++nt) {
    const int n    = nt * 16 + l16;
    const float bb = b2[n];
#pragma unroll
    for (int i = 0; i < 8; ++i) {
      float v = acc[nt][i] + bb;
      v = v > 0.f ? v : 0.f;
      const int grow = row0 + m0 + 8 * hl + i;
      if (HIST) {
        atomicAdd(&seg_sum[(size_t)cidx[i] * 128 + n], v);
      } else {
        out_h[(size_t)grow * 128 + n] = v;
      }
    }
  }
  if (HIST && tid < 64) atomicAdd(&seg_cnt[seg_idx[row0 + tid]], 1.0f);
}

// ---------------------------------------------------------------------------
// Kernel 2: segment-mean finalize + 2-layer card MLP (133 (pad 160) -> 128 -> 128).
// ---------------------------------------------------------------------------
__global__ __launch_bounds__(128) void card_mlp_kernel(
    const float* __restrict__ dense,     // [U][5]
    const float* __restrict__ seg_sum,   // [U][128]
    const float* __restrict__ seg_cnt,   // [U]
    const bf16_t* __restrict__ Wp1,      // packed, padK=160
    const float* __restrict__ bc1,
    const bf16_t* __restrict__ Wp2, const float* __restrict__ bc2,
    float* __restrict__ card_h, int U) {
  __shared__ bf16_t sBuf[64 * 160];  // 20 KB: X tile (64x160) then H1 (64x128)
  const int tid  = threadIdx.x;
  const int row0 = blockIdx.x * 64;

  // Stage: cols 0..4 dense, 5..132 = seg_sum/max(cnt,1), 133..159 zero pad.
  {
    const int r = tid >> 1, half = tid & 1;
    const int u = row0 + r;
    const bool valid = u < U;
    float inv = 0.f;
    if (valid) inv = 1.0f / fmaxf(seg_cnt[u], 1.0f);
    for (int c = half * 80; c < half * 80 + 80; ++c) {
      float v = 0.f;
      if (valid) {
        if (c < 5)        v = dense[(size_t)u * 5 + c];
        else if (c < 133) v = seg_sum[(size_t)u * 128 + (c - 5)] * inv;
      }
      sBuf[r * 160 + c] = (bf16_t)v;
    }
  }
  __syncthreads();

  const int lane = tid & 31;
  const int hl   = lane >> 4;
  const int l16  = lane & 15;
  const int m0   = (tid >> 5) * 16;

  v8f acc[8];
  wave_gemm_bf16(sBuf, 160, m0, 160, Wp1, acc);

  bf16_t h1[8][8];
#pragma unroll
  for (int nt = 0; nt < 8; ++nt) {
    const float bb = bc1[nt * 16 + l16];
#pragma unroll
    for (int i = 0; i < 8; ++i) {
      float v = acc[nt][i] + bb;
      h1[nt][i] = (bf16_t)(v > 0.f ? v : 0.f);
    }
  }
  __syncthreads();
#pragma unroll
  for (int nt = 0; nt < 8; ++nt) {
    const int n = nt * 16 + l16;
#pragma unroll
    for (int i = 0; i < 8; ++i) sBuf[(m0 + 8 * hl + i) * 128 + n] = h1[nt][i];
  }
  __syncthreads();

  wave_gemm_bf16(sBuf, 128, m0, 128, Wp2, acc);
#pragma unroll
  for (int nt = 0; nt < 8; ++nt) {
    const int n    = nt * 16 + l16;
    const float bb = bc2[n];
#pragma unroll
    for (int i = 0; i < 8; ++i) {
      const int u = row0 + m0 + 8 * hl + i;
      if (u < U) {
        float v = acc[nt][i] + bb;
        card_h[(size_t)u * 128 + n] = v > 0.f ? v : 0.f;
      }
    }
  }
}

// ---------------------------------------------------------------------------
// Kernel 3: head. 64 threads = 2 waves, 32 rows/block. Gather concat tile,
// bf16 WMMA layer (K=256), then 128->1 dot in f32 + shfl_xor reduce.
// ---------------------------------------------------------------------------
__global__ __launch_bounds__(64) void head_kernel(
    const float* __restrict__ target_h,  // [B][128]
    const float* __restrict__ card_h,    // [U][128]
    const int* __restrict__ tcidx,       // [B]
    const bf16_t* __restrict__ Wp1,      // packed, K=256
    const float* __restrict__ bh1,
    const float* __restrict__ Wh2,       // [128]
    const float* __restrict__ bh2,       // [1]
    float* __restrict__ out) {
  __shared__ bf16_t sX[32 * 256];  // 16 KB
  const int tid  = threadIdx.x;
  const int row0 = blockIdx.x * 32;

  {
    const int r = tid >> 1, half = tid & 1;
    const int b = row0 + r;
    const float* src = (half == 0) ? (target_h + (size_t)b * 128)
                                   : (card_h + (size_t)tcidx[b] * 128);
    const float4* s = (const float4*)src;
    v4bf* d = (v4bf*)(sX + r * 256 + half * 128);
#pragma unroll
    for (int i = 0; i < 32; ++i) {
      float4 f = s[i];
      v4bf o = {(bf16_t)f.x, (bf16_t)f.y, (bf16_t)f.z, (bf16_t)f.w};
      d[i] = o;
    }
  }
  __syncthreads();

  const int lane = tid & 31;
  const int hl   = lane >> 4;
  const int l16  = lane & 15;
  const int m0   = (tid >> 5) * 16;

  v8f acc[8];
  wave_gemm_bf16(sX, 256, m0, 256, Wp1, acc);

  float part[8] = {0.f, 0.f, 0.f, 0.f, 0.f, 0.f, 0.f, 0.f};
#pragma unroll
  for (int nt = 0; nt < 8; ++nt) {
    const int n    = nt * 16 + l16;
    const float bb = bh1[n];
    const float w2 = Wh2[n];
#pragma unroll
    for (int i = 0; i < 8; ++i) {
      float v = acc[nt][i] + bb;
      v = v > 0.f ? v : 0.f;
      part[i] += v * w2;
    }
  }
  // Reduce across the 16 N-lanes of each half-wave (xor masks keep hl fixed).
#pragma unroll
  for (int m = 1; m < 16; m <<= 1) {
#pragma unroll
    for (int i = 0; i < 8; ++i) part[i] += __shfl_xor(part[i], m, 32);
  }
  if (l16 == 0) {
    const float bias = bh2[0];
#pragma unroll
    for (int i = 0; i < 8; ++i) out[row0 + m0 + 8 * hl + i] = part[i] + bias;
  }
}

// ---------------------------------------------------------------------------
extern "C" void kernel_launch(void* const* d_in, const int* in_sizes, int n_in,
                              void* d_out, int out_size, void* d_ws, size_t ws_size,
                              hipStream_t stream) {
  const float* target_x = (const float*)d_in[0];
  const float* hist_x   = (const float*)d_in[1];
  const int*   hidx     = (const int*)d_in[2];
  const int*   tidx     = (const int*)d_in[3];
  const float* dense    = (const float*)d_in[4];
  const float* W_t1 = (const float*)d_in[5];  const float* b_t1 = (const float*)d_in[6];
  const float* W_t2 = (const float*)d_in[7];  const float* b_t2 = (const float*)d_in[8];
  const float* W_c1 = (const float*)d_in[9];  const float* b_c1 = (const float*)d_in[10];
  const float* W_c2 = (const float*)d_in[11]; const float* b_c2 = (const float*)d_in[12];
  const float* W_h1 = (const float*)d_in[13]; const float* b_h1 = (const float*)d_in[14];
  const float* W_h2 = (const float*)d_in[15]; const float* b_h2 = (const float*)d_in[16];

  const int B = in_sizes[0] / 64;
  const int H = in_sizes[1] / 64;
  const int U = in_sizes[4] / 5;

  char* ws = (char*)d_ws;
  size_t off = 0;
  auto alloc = [&](size_t bytes) -> void* {
    void* p = ws + off;
    off = (off + bytes + 255) & ~(size_t)255;
    return p;
  };
  float* target_h = (float*)alloc((size_t)B * 128 * sizeof(float));
  float* hist_sum = (float*)alloc((size_t)U * 128 * sizeof(float));
  float* hist_cnt = (float*)alloc((size_t)U * sizeof(float));
  float* card_h   = (float*)alloc((size_t)U * 128 * sizeof(float));
  bf16_t* Wp_t1 = (bf16_t*)alloc((size_t)64  * 128 * sizeof(bf16_t));
  bf16_t* Wp_t2 = (bf16_t*)alloc((size_t)128 * 128 * sizeof(bf16_t));
  bf16_t* Wp_c1 = (bf16_t*)alloc((size_t)160 * 128 * sizeof(bf16_t));
  bf16_t* Wp_c2 = (bf16_t*)alloc((size_t)128 * 128 * sizeof(bf16_t));
  bf16_t* Wp_h1 = (bf16_t*)alloc((size_t)256 * 128 * sizeof(bf16_t));

  hipMemsetAsync(hist_sum, 0, (size_t)U * 128 * sizeof(float), stream);
  hipMemsetAsync(hist_cnt, 0, (size_t)U * sizeof(float), stream);

  pack_w_kernel<<<(64  * 128 + 255) / 256, 256, 0, stream>>>(W_t1, Wp_t1, 64,  64);
  pack_w_kernel<<<(128 * 128 + 255) / 256, 256, 0, stream>>>(W_t2, Wp_t2, 128, 128);
  pack_w_kernel<<<(160 * 128 + 255) / 256, 256, 0, stream>>>(W_c1, Wp_c1, 133, 160);
  pack_w_kernel<<<(128 * 128 + 255) / 256, 256, 0, stream>>>(W_c2, Wp_c2, 128, 128);
  pack_w_kernel<<<(256 * 128 + 255) / 256, 256, 0, stream>>>(W_h1, Wp_h1, 256, 256);

  txn_mlp_kernel<false><<<B / 64, 128, 0, stream>>>(
      target_x, Wp_t1, b_t1, Wp_t2, b_t2, target_h, nullptr, nullptr, nullptr);
  txn_mlp_kernel<true><<<H / 64, 128, 0, stream>>>(
      hist_x, Wp_t1, b_t1, Wp_t2, b_t2, nullptr, hist_sum, hist_cnt, hidx);

  card_mlp_kernel<<<(U + 63) / 64, 128, 0, stream>>>(
      dense, hist_sum, hist_cnt, Wp_c1, b_c1, Wp_c2, b_c2, card_h, U);

  head_kernel<<<B / 32, 64, 0, stream>>>(
      target_h, card_h, tidx, Wp_h1, b_h1, W_h2, b_h2, (float*)d_out);
}
